// LTCEcgClassifier_73297911873564
// MI455X (gfx1250) — matrix operand order repaired and checked by reading, not compiled, not run on hardware
//
#include <hip/hip_runtime.h>
#include <stdint.h>

// ---------------------------------------------------------------------------
// LTC ECG classifier for MI455X (gfx1250, wave32, WMMA)
//   B=128, T=2048, D=H=256, L=2, C=5, DT=0.1
// Persistent-kernel sequential scan, bf16 WMMA (f32 accum), async LDS staging.
// Each workgroup owns one 16-row m-tile; all A-operands are async-staged to
// LDS (16 rows only) and consumed as ds_load_b128 fragments.
// ---------------------------------------------------------------------------

typedef __bf16 bf16_t;
typedef __attribute__((ext_vector_type(16))) bf16_t v16bf;
typedef __attribute__((ext_vector_type(8)))  bf16_t v8bf;
typedef __attribute__((ext_vector_type(8)))  float  v8f;
typedef unsigned short ushort;
typedef unsigned int   uint;

#define B_   128
#define T_   2048
#define D_   256
#define H_   256
#define NWG  16
#define SCAN_THREADS 512            // 16 waves per WG -> 256 waves total
#define ACT_STRIDE   536            // ushorts/row: 16B aligned, conflict-free

// ---- bf16 helpers ---------------------------------------------------------
__device__ __forceinline__ ushort f2bf(float f) {
  union { float f; uint u; } v; v.f = f;
  uint r = (v.u + 0x7FFFu + ((v.u >> 16) & 1u)) >> 16;   // RNE
  return (ushort)r;
}

// ---- WMMA wrappers --------------------------------------------------------
__device__ __forceinline__ v8f wmma_bf16(v16bf a, v16bf b, v8f c) {
  return __builtin_amdgcn_wmma_f32_16x16x32_bf16(
      /*neg_a=*/false, a, /*neg_b=*/false, b,
      /*c_mod=*/(short)0, c, /*reuse_a=*/false, /*reuse_b=*/false);
}

// A-fragment (16x32, 16-bit) from a row-major bf16 matrix:
//   lane<16 : row=lane,   K = {kb..kb+7, kb+16..kb+23} with kb = kt*32
//   lane>=16: row=lane-16, K = {kb+8..kb+15, kb+24..kb+31}
__device__ __forceinline__ v16bf a_frag(const ushort* base, int stride,
                                        int lane, int kt) {
  const int m  = lane & 15;
  const int kb = kt * 32 + ((lane >> 4) << 3);
  const ushort* p = base + m * stride + kb;
  v8bf lo = *(const v8bf*)p;
  v8bf hi = *(const v8bf*)(p + 16);
  v16bf a;
#pragma unroll
  for (int i = 0; i < 8; ++i) { a[i] = lo[i]; a[i + 8] = hi[i]; }
  return a;
}

// B-fragment from pre-packed weights: [kt][nt][lane][16] contiguous (32B/lane)
__device__ __forceinline__ v16bf b_frag(const ushort* wp, int kt, int nt, int lane) {
  return *(const v16bf*)(wp + (((kt << 4) + nt) * 32 + lane) * 16);
}

// ---- CDNA5 async global->LDS copy (ASYNCcnt tracked) ----------------------
__device__ __forceinline__ void async_copy16(ushort* ldsDst, const ushort* gSrc) {
  unsigned la = (unsigned)(uintptr_t)ldsDst;   // low 32 bits = LDS offset
  asm volatile("global_load_async_to_lds_b128 %0, %1, off"
               :: "v"(la), "v"(gSrc) : "memory");
}
__device__ __forceinline__ void async_wait0() {
  asm volatile("s_wait_asynccnt 0x0" ::: "memory");
}

// Stage rows [r0, r0+16) of [m0 | m1] (two 128x256 bf16 matrices) into
// LDS act[16][512]+pad.  1024 16B chunks / 512 threads = 2 per thread.
__device__ __forceinline__ void stage_pair16(ushort* act, const ushort* m0,
                                             const ushort* m1, int r0) {
#pragma unroll
  for (int i = 0; i < 2; ++i) {
    int c   = threadIdx.x + i * SCAN_THREADS;  // [0, 1024)
    int mat = c >> 9;
    int r   = (c >> 5) & 15;
    int seg = c & 31;
    const ushort* src = (mat ? m1 : m0) + (r0 + r) * 256 + seg * 8;
    ushort*       dst = act + r * ACT_STRIDE + mat * 256 + seg * 8;
    async_copy16(dst, src);
  }
  async_wait0();
}

// ---- inter-workgroup barrier (16 co-resident WGs, sense-reversing) --------
__device__ __forceinline__ void grid_barrier(uint* cnt, uint* gen) {
  __threadfence();
  __syncthreads();
  if (threadIdx.x == 0) {
    uint g = __hip_atomic_load(gen, __ATOMIC_ACQUIRE, __HIP_MEMORY_SCOPE_AGENT);
    uint a = __hip_atomic_fetch_add(cnt, 1u, __ATOMIC_ACQ_REL, __HIP_MEMORY_SCOPE_AGENT);
    if (a == NWG - 1) {
      __hip_atomic_store(cnt, 0u, __ATOMIC_RELAXED, __HIP_MEMORY_SCOPE_AGENT);
      __hip_atomic_fetch_add(gen, 1u, __ATOMIC_ACQ_REL, __HIP_MEMORY_SCOPE_AGENT);
    } else {
      while (__hip_atomic_load(gen, __ATOMIC_ACQUIRE, __HIP_MEMORY_SCOPE_AGENT) == g)
        __builtin_amdgcn_s_sleep(1);
    }
  }
  __syncthreads();
}

// ---- stage 1: g = act( [inp|h] @ W1 + b1 ), both tau & f paths ------------
// 256 tile jobs across 256 waves; K = 512 -> 16 WMMAs.
// LDS act holds this WG's 16 rows of [inp|h].
__device__ __forceinline__ void stage1(int l, const ushort* act, const ushort* wpk,
                                       const float* taub1, const float* fb1,
                                       ushort* gt, ushort* gf, int gw, int lane) {
  const int path = gw >> 7;          // 0 = tau, 1 = f
  const int rem  = gw & 127;
  const int mt   = rem >> 4;         // 8 m-tiles (== blockIdx & 7)
  const int nt   = rem & 15;         // 16 n-tiles
  const ushort* wp = wpk + l * 393216 + path * 131072;
  v8f acc = {};
#pragma unroll
  for (int kt = 0; kt < 16; ++kt)
    acc = wmma_bf16(a_frag(act, ACT_STRIDE, lane, kt),
                    b_frag(wp, kt, nt, lane), acc);
  const int col    = nt * 16 + (lane & 15);
  const float bias = (path ? fb1 : taub1)[l * H_ + col];
  ushort* dst = path ? gf : gt;
#pragma unroll
  for (int r = 0; r < 8; ++r) {
    int row = mt * 16 + r + ((lane >> 4) << 3);
    float v = acc[r] + bias;
    v = path ? tanhf(v) : fmaxf(v, 0.0f);
    dst[row * H_ + col] = f2bf(v);
  }
}

// ---- stage 2: tau/f second GEMMs + fused h update -------------------------
// 128 jobs on waves 0..127; each wave does BOTH paths for one (m,n) tile.
// LDS act holds this WG's 16 rows of [gt | gf]: tau frags kt 0..7, f kt 8..15.
__device__ __forceinline__ void stage2(int l, const ushort* act,
                                       const ushort* wpk, const float* taub2,
                                       const float* fb2, float* hF, ushort* hB,
                                       int gw, int lane) {
  if (gw >= 128) return;             // wave-uniform: EXEC stays full
  const int mt = gw >> 4;
  const int nt = gw & 15;
  const ushort* w2t = wpk + l * 393216 + 262144;
  const ushort* w2f = wpk + l * 393216 + 327680;
  v8f at = {}, af = {};
#pragma unroll
  for (int kt = 0; kt < 8; ++kt)
    at = wmma_bf16(a_frag(act, ACT_STRIDE, lane, kt),     b_frag(w2t, kt, nt, lane), at);
#pragma unroll
  for (int kt = 0; kt < 8; ++kt)
    af = wmma_bf16(a_frag(act, ACT_STRIDE, lane, kt + 8), b_frag(w2f, kt, nt, lane), af);
  const int col = nt * 16 + (lane & 15);
  const float bt = taub2[l * H_ + col];
  const float bf = fb2[l * H_ + col];
#pragma unroll
  for (int r = 0; r < 8; ++r) {
    int row = mt * 16 + r + ((lane >> 4) << 3);
    float tau = 1.0f / (1.0f + __expf(-(at[r] + bt))) + 0.1f;
    float f   = tanhf(af[r] + bf);
    float h   = hF[row * H_ + col];
    float hn  = h + 0.1f * (f - h) / (tau + 1e-6f);
    hF[row * H_ + col] = hn;
    hB[row * H_ + col] = f2bf(hn);
  }
}

// ---- persistent sequential scan -------------------------------------------
__global__ __launch_bounds__(SCAN_THREADS)
void ltc_scan_kernel(const ushort* __restrict__ xbf, const ushort* __restrict__ wpk,
                     const float* taub1, const float* taub2,
                     const float* fb1, const float* fb2,
                     float* h0f, float* h1f, ushort* h0b, ushort* h1b,
                     ushort* gt, ushort* gf, uint* bar) {
  extern __shared__ ushort act[];    // 16 x ACT_STRIDE bf16 (17152 B)
  const int lane = threadIdx.x & 31;
  const int gw   = blockIdx.x * (SCAN_THREADS / 32) + (threadIdx.x >> 5);
  const int r0   = (blockIdx.x & 7) * 16;   // this WG's m-rows
  uint* cnt = bar; uint* gen = bar + 1;

  for (int t = 0; t < T_; ++t) {
    // ---- layer 0 ----
    stage_pair16(act, xbf + (size_t)t * (B_ * D_), h0b, r0);   // A = [x_t | h0]
    __syncthreads();
    stage1(0, act, wpk, taub1, fb1, gt, gf, gw, lane);
    grid_barrier(cnt, gen);
    stage_pair16(act, gt, gf, r0);                             // A = [gt | gf]
    __syncthreads();
    stage2(0, act, wpk, taub2, fb2, h0f, h0b, gw, lane);
    grid_barrier(cnt, gen);
    // ---- layer 1 ----
    stage_pair16(act, h0b, h1b, r0);                           // A = [h0' | h1]
    __syncthreads();
    stage1(1, act, wpk, taub1, fb1, gt, gf, gw, lane);
    grid_barrier(cnt, gen);
    stage_pair16(act, gt, gf, r0);
    __syncthreads();
    stage2(1, act, wpk, taub2, fb2, h1f, h1b, gw, lane);
    grid_barrier(cnt, gen);
  }
}

// ---- one-time weight pack: fp32 -> bf16, WMMA B-fragment layout -----------
// packed elem addr = ((kt*16+nt)*32 + lane)*16 + t ;
// source (k,n): k = kt*32 + (lane>=16)*8 + (t&7) + (t>=8)*16 ; n = nt*16 + (lane&15)
__global__ __launch_bounds__(256)
void pack_weights_kernel(const float* tau_w1, const float* f_w1,
                         const float* tau_w2, const float* f_w2, ushort* wpk) {
  int tid = blockIdx.x * 256 + threadIdx.x;          // [0, 786432)
  int l = tid / 393216;
  int r = tid - l * 393216;
  const float* src; int e; int dstoff = l * 393216;
  if (r < 131072)      { src = tau_w1 + l * 131072; e = r;          dstoff += 0;      }
  else if (r < 262144) { src = f_w1   + l * 131072; e = r - 131072; dstoff += 131072; }
  else if (r < 327680) { src = tau_w2 + l * 65536;  e = r - 262144; dstoff += 262144; }
  else                 { src = f_w2   + l * 65536;  e = r - 327680; dstoff += 327680; }
  int tt   = e & 15;
  int lane = (e >> 4) & 31;
  int nt   = (e >> 9) & 15;
  int kt   = e >> 13;
  int k = kt * 32 + ((lane >> 4) << 3) + (tt & 7) + ((tt >> 3) << 4);
  int n = nt * 16 + (lane & 15);
  wpk[dstoff + e] = f2bf(src[k * 256 + n]);
}

// ---- one-time x transpose+convert: x[b][t][d] fp32 -> xbf[t][b][d] bf16 ---
__global__ __launch_bounds__(256)
void convert_x_kernel(const float* __restrict__ x, ushort* __restrict__ xbf) {
  size_t tid = (size_t)blockIdx.x * 256 + threadIdx.x;   // < 67108864
  int d = (int)(tid & 255);
  int b = (int)((tid >> 8) & 127);
  int t = (int)(tid >> 15);
  xbf[tid] = f2bf(x[(size_t)b * (T_ * D_) + (size_t)t * D_ + d]);
}

// ---- tiny classifier head: 256 -> 128 -> 64 -> 5 --------------------------
__global__ __launch_bounds__(128)
void classifier_kernel(const float* __restrict__ h1f,
                       const float* w1, const float* b1,
                       const float* w2, const float* b2,
                       const float* w3, const float* b3, float* out) {
  __shared__ float z1[128];
  __shared__ float z2[64];
  const float* hr = h1f + blockIdx.x * H_;
  int j = threadIdx.x;
  float s = b1[j];
  for (int k = 0; k < H_; ++k) s += hr[k] * w1[k * 128 + j];
  z1[j] = fmaxf(s, 0.0f);
  __syncthreads();
  if (j < 64) {
    float s2 = b2[j];
    for (int k = 0; k < 128; ++k) s2 += z1[k] * w2[k * 64 + j];
    z2[j] = fmaxf(s2, 0.0f);
  }
  __syncthreads();
  if (j < 5) {
    float s3 = b3[j];
    for (int k = 0; k < 64; ++k) s3 += z2[k] * w3[k * 5 + j];
    out[blockIdx.x * 5 + j] = s3;
  }
}

// ---------------------------------------------------------------------------
extern "C" void kernel_launch(void* const* d_in, const int* in_sizes, int n_in,
                              void* d_out, int out_size, void* d_ws, size_t ws_size,
                              hipStream_t stream) {
  (void)in_sizes; (void)n_in; (void)out_size; (void)ws_size;
  const float* x      = (const float*)d_in[0];
  const float* tau_w1 = (const float*)d_in[1];
  const float* tau_b1 = (const float*)d_in[2];
  const float* tau_w2 = (const float*)d_in[3];
  const float* tau_b2 = (const float*)d_in[4];
  const float* f_w1   = (const float*)d_in[5];
  const float* f_b1   = (const float*)d_in[6];
  const float* f_w2   = (const float*)d_in[7];
  const float* f_b2   = (const float*)d_in[8];
  const float* clf_w1 = (const float*)d_in[9];
  const float* clf_b1 = (const float*)d_in[10];
  const float* clf_w2 = (const float*)d_in[11];
  const float* clf_b2 = (const float*)d_in[12];
  const float* clf_w3 = (const float*)d_in[13];
  const float* clf_b3 = (const float*)d_in[14];

  // workspace layout (bytes):
  //   [0,256)        barrier cnt/gen
  //   [256,..)       h0 fp32 (128K), h1 fp32 (128K), h0 bf16 (64K), h1 bf16 (64K)
  //   gt bf16 (64K), gf bf16 (64K)
  //   packed weights bf16 (1.5M), xbf bf16 [t][b][d] (128M)
  char*   ws  = (char*)d_ws;
  uint*   bar = (uint*)ws;
  float*  h0f = (float*)(ws + 256);
  float*  h1f = (float*)(ws + 131328);
  ushort* h0b = (ushort*)(ws + 262400);
  ushort* h1b = (ushort*)(ws + 327936);
  ushort* gt  = (ushort*)(ws + 393472);
  ushort* gf  = (ushort*)(ws + 459008);
  ushort* wpk = (ushort*)(ws + 524544);
  ushort* xbf = (ushort*)(ws + 2097408);

  hipMemsetAsync(ws, 0, 393472, stream);   // zero barriers + h state
  pack_weights_kernel<<<3072, 256, 0, stream>>>(tau_w1, f_w1, tau_w2, f_w2, wpk);
  convert_x_kernel<<<262144, 256, 0, stream>>>(x, xbf);
  ltc_scan_kernel<<<NWG, SCAN_THREADS, 16 * ACT_STRIDE * 2, stream>>>(
      xbf, wpk, tau_b1, tau_b2, f_b1, f_b2, h0f, h1f, h0b, h1b, gt, gf, bar);
  classifier_kernel<<<128, 128, 0, stream>>>(h1f, clf_w1, clf_b1, clf_w2,
                                             clf_b2, clf_w3, clf_b3, (float*)d_out);
}
